// BCNNResNet_55542517072356
// MI455X (gfx1250) — matrix operand order repaired and checked by reading, not compiled, not run on hardware
//
#include <hip/hip_runtime.h>

typedef __attribute__((ext_vector_type(16))) __bf16 v16bf;
typedef __attribute__((ext_vector_type(8)))  float  v8f;
typedef __attribute__((ext_vector_type(4)))  float  v4f;

typedef __attribute__((ext_vector_type(4))) unsigned int tdm_v4u;
typedef __attribute__((ext_vector_type(8))) int          tdm_v8i;
typedef __attribute__((ext_vector_type(4))) int          tdm_v4i;

#define NB 32      // batch
#define ND 2048    // feature dim
#define NC 30      // classes
#define KTILES (ND / 32)        // 64 k-steps of 32
#define MROWS (NC * ND)         // 61440 rows of flat W
#define MTILE 128               // rows per workgroup (4 waves x 32 rows)
#define NWG   (MROWS / MTILE)   // 480
#define NSEG  128               // partials per (b,c): 16 tiles/class * 4 waves * 2 lane-halves
#define SPITCH 36               // LDS slab row pitch in floats (32 data + 4 pad: conflict-free)

#if defined(__has_builtin)
#if __has_builtin(__builtin_amdgcn_tensor_load_to_lds)
#define USE_TDM 1
#endif
#endif

// ---------------- prep: g[b,d] = sign(f)*sqrt(|f|) * norm_b^{-1/2} ----------------
__global__ void __launch_bounds__(256) bcnn_prep(const float* __restrict__ feat,
                                                 float* __restrict__ gF) {
    __shared__ float red[256];
    const int b = blockIdx.x;
    const float* f = feat + b * ND;
    float s = 0.f;
    for (int d = threadIdx.x; d < ND; d += 256) s += fabsf(f[d]);
    red[threadIdx.x] = s;
    __syncthreads();
    for (int off = 128; off > 0; off >>= 1) {
        if (threadIdx.x < off) red[threadIdx.x] += red[threadIdx.x + off];
        __syncthreads();
    }
    const float S = red[0];
    // ||y||^2 = S^2 + D^2 * EPS_SQRT exactly (sum over i,j of |fi*fj| + eps)
    float nrm = sqrtf(S * S + (float)ND * (float)ND * 1e-10f);
    nrm = fmaxf(nrm, 1e-12f);
    const float sb = rsqrtf(nrm);   // norm^{-1/2}: applied to both sides of the quadratic form
    for (int d = threadIdx.x; d < ND; d += 256) {
        float x = f[d];
        gF[b * ND + d] = copysignf(sqrtf(fabsf(x)), x) * sb;
    }
}

// ---------------- pack: B operand in per-lane WMMA register order, contiguous k ----------------
// Flat layout: Bp[((kt*2 + nt)*32 + lane)*16 + t]
// 32x16 B in this code's contiguous k order: lane l -> N = nt*16 + (l&15); lanes>=16 hold K = 16..31.
// element t (VGPR t/2, half t%2) -> K = kt*32 + 16*(l>>4) + t.
__global__ void __launch_bounds__(256) bcnn_pack(const float* __restrict__ gF,
                                                 __bf16* __restrict__ Bp) {
    const int tid = blockIdx.x * 256 + threadIdx.x;   // 65536 total
    const int t  = tid & 15;
    const int l  = (tid >> 4) & 31;
    const int nt = (tid >> 9) & 1;
    const int kt = tid >> 10;
    const int b  = nt * 16 + (l & 15);
    const int k  = kt * 32 + 16 * (l >> 4) + t;
    Bp[tid] = (__bf16)gF[b * ND + k];
}

#ifdef USE_TDM
// Issue a TDM 2D tile load: 128 rows x 32 f32, row stride 2048 f32 in memory,
// destination LDS padded to SPITCH (pad_interval=32 DW -> code 4, pad_amount=4 DW -> code 3).
__device__ __forceinline__ void tdm_issue(unsigned lds_off, unsigned long long ga) {
    tdm_v4u g0;
    tdm_v8i g1;
    tdm_v4i g2 = {0, 0, 0, 0}, g3 = {0, 0, 0, 0};
    g0[0] = 1u;                                          // count=1, no gather
    g0[1] = lds_off;                                     // lds_addr (bytes)
    g0[2] = (unsigned)(ga & 0xFFFFFFFFull);              // global_addr[31:0]
    g0[3] = (unsigned)((ga >> 32) & 0x01FFFFFFull)       // global_addr[56:32]
          | 0x80000000u;                                 // type=2 ("image")
    g1[0] = (int)0x07120000;   // data_size=4B(2), pad_enable=1, pad_interval=4(32DW), pad_amount=3(4DW)
    g1[1] = (int)(2048u << 16);                          // tensor_dim0 = 2048 (bits 79:48)
    g1[2] = (int)(128u  << 16);                          // tensor_dim1 = 128 (bits 111:80)
    g1[3] = (int)(32u   << 16);                          // tile_dim0 = 32 (bits 127:112)
    g1[4] = 128;                                         // tile_dim1 = 128, tile_dim2 = 0
    g1[5] = 2048;                                        // tensor_dim0_stride lo32
    g1[6] = 0;                                           // stride hi, dim1_stride lo
    g1[7] = 0;
#if __has_include(<hip/amd_detail/amd_gfx1250_TDM.h>)
    tdm_v8i g4 = {0, 0, 0, 0, 0, 0, 0, 0};               // 6-arg toolchain variant
    __builtin_amdgcn_tensor_load_to_lds(g0, g1, g2, g3, g4, 0);
#else
    __builtin_amdgcn_tensor_load_to_lds(g0, g1, g2, g3, 0);
#endif
}
#endif

__device__ __forceinline__ void stage_slab(const float* __restrict__ W, int rowBase0,
                                           int kt, float* sbuf, int wave) {
#ifdef USE_TDM
    if (wave == 0) {
        unsigned lds_off = (unsigned)(uintptr_t)sbuf;    // low 32 bits = LDS byte offset
        unsigned long long ga = (unsigned long long)(uintptr_t)W +
            ((unsigned long long)rowBase0 * (unsigned long long)ND +
             (unsigned long long)kt * 32ull) * 4ull;
        tdm_issue(lds_off, ga);
    }
#else
    // cooperative synchronous copy: thread t copies row t (32 floats) at SPITCH pitch
    const int t = threadIdx.x;
    const v4f* src = (const v4f*)(W + (size_t)(rowBase0 + t) * ND + (size_t)kt * 32);
    v4f* dst = (v4f*)(sbuf + t * SPITCH);
    #pragma unroll
    for (int i = 0; i < 8; ++i) dst[i] = src[i];
    (void)wave;
#endif
}

// ---------------- gemm: V = W[61440,2048] @ G[2048,32], fused g^T * V contraction ----------------
__global__ void __launch_bounds__(128) bcnn_gemm(const float* __restrict__ W,
                                                 const __bf16* __restrict__ Bp,
                                                 const float* __restrict__ gF,
                                                 float* __restrict__ partial) {
    __shared__ float slab[2][MTILE * SPITCH];            // 2 x 18 KB, TDM double buffer

    const int lane = threadIdx.x & 31;
    const int wave = threadIdx.x >> 5;
    const int lm   = lane & 15;
    const int h    = lane >> 4;
    const int rowBase0 = blockIdx.x * MTILE;             // workgroup's first W row
    const int rowBase  = rowBase0 + wave * 32;           // this wave's first W row

    v8f acc00 = {}, acc01 = {}, acc10 = {}, acc11 = {};

    // prologue: fill both buffers
    stage_slab(W, rowBase0, 0, &slab[0][0], wave);
    stage_slab(W, rowBase0, 1, &slab[1][0], wave);

    for (int kt = 0; kt < KTILES; ++kt) {
        const int bb = kt & 1;
#ifdef USE_TDM
        if (wave == 0) {
            if (kt < KTILES - 1) __builtin_amdgcn_s_wait_tensorcnt(1);
            else                 __builtin_amdgcn_s_wait_tensorcnt(0);
        }
#endif
        __syncthreads();   // slab[bb] ready for all waves

        // A fragments from LDS (contiguous k order, same as Bp):
        // lanes 0-15: row M=lm, K in {0..15}; lanes 16-31: K in {16..31}
        const float* r0 = &slab[bb][0] + (wave * 32 + lm) * SPITCH;        // mt = 0
        const float* r1 = &slab[bb][0] + (wave * 32 + 16 + lm) * SPITCH;   // mt = 1
        const v4f* p0 = (const v4f*)(r0 + 16 * h);
        const v4f* p1 = (const v4f*)(r1 + 16 * h);
        v4f x0 = p0[0], x1 = p0[1], x2 = p0[2], x3 = p0[3];  // K: 16h..16h+15
        v4f y0 = p1[0], y1 = p1[1], y2 = p1[2], y3 = p1[3];

        v16bf a0, a1;
        #pragma unroll
        for (int t = 0; t < 4; ++t) {
            a0[t]      = (__bf16)x0[t];  a0[4 + t]  = (__bf16)x1[t];
            a0[8 + t]  = (__bf16)x2[t];  a0[12 + t] = (__bf16)x3[t];
            a1[t]      = (__bf16)y0[t];  a1[4 + t]  = (__bf16)y1[t];
            a1[8 + t]  = (__bf16)y2[t];  a1[12 + t] = (__bf16)y3[t];
        }

        const v16bf b0 = *(const v16bf*)(Bp + ((size_t)(kt * 2 + 0) * 32 + lane) * 16);
        const v16bf b1 = *(const v16bf*)(Bp + ((size_t)(kt * 2 + 1) * 32 + lane) * 16);

        acc00 = __builtin_amdgcn_wmma_f32_16x16x32_bf16(false, a0, false, b0, (short)0, acc00, false, false);
        acc01 = __builtin_amdgcn_wmma_f32_16x16x32_bf16(false, a0, false, b1, (short)0, acc01, false, false);
        acc10 = __builtin_amdgcn_wmma_f32_16x16x32_bf16(false, a1, false, b0, (short)0, acc10, false, false);
        acc11 = __builtin_amdgcn_wmma_f32_16x16x32_bf16(false, a1, false, b1, (short)0, acc11, false, false);

        __syncthreads();   // everyone done reading slab[bb] before it is refilled
        if (kt + 2 < KTILES) stage_slab(W, rowBase0, kt + 2, &slab[bb][0], wave);
    }

    // Epilogue: out[b,c] partial = sum_i g[b,i] * V[c*D+i, b].
    // C/D layout: lane holds N = nt*16+lm; VGPR r is row (r + 8h) of the 16-row subtile.
    const int c     = rowBase / ND;   // MTILE=128 divides D: tiles never cross a class
    const int iBase = rowBase % ND;
    float pa = 0.f, pb = 0.f;
    #pragma unroll
    for (int r = 0; r < 8; ++r) {
        const int i0 = iBase + r + 8 * h;        // mt = 0 rows
        const int i1 = iBase + 16 + r + 8 * h;   // mt = 1 rows
        pa += gF[lm * ND + i0]        * acc00[r] + gF[lm * ND + i1]        * acc10[r];
        pb += gF[(16 + lm) * ND + i0] * acc01[r] + gF[(16 + lm) * ND + i1] * acc11[r];
    }
    const int tIC = blockIdx.x & 15;                 // tile index within class
    const int seg = ((tIC * 4 + wave) << 1) | h;     // 0..127, unique per (b,c) contributor
    partial[((size_t)(c * NB + lm) * NSEG) + seg]      = pa;
    partial[((size_t)(c * NB + 16 + lm) * NSEG) + seg] = pb;
}

// ---------------- finalize: reduce partials, add bias ----------------
__global__ void __launch_bounds__(256) bcnn_finalize(const float* __restrict__ partial,
                                                     const float* __restrict__ bias,
                                                     float* __restrict__ out) {
    const int tid = blockIdx.x * 256 + threadIdx.x;
    if (tid >= NB * NC) return;
    const int b = tid / NC, c = tid % NC;
    const float* p = partial + (size_t)(c * NB + b) * NSEG;
    float s = 0.f;
    #pragma unroll 8
    for (int i = 0; i < NSEG; ++i) s += p[i];
    out[tid] = s + bias[c];
}

extern "C" void kernel_launch(void* const* d_in, const int* in_sizes, int n_in,
                              void* d_out, int out_size, void* d_ws, size_t ws_size,
                              hipStream_t stream) {
    const float* feat = (const float*)d_in[0];   // [32, 2048]
    const float* W    = (const float*)d_in[1];   // [30, 2048*2048] == [61440, 2048]
    const float* bias = (const float*)d_in[2];   // [30]
    float* out = (float*)d_out;                  // [32, 30]

    char* ws = (char*)d_ws;
    float*  gF      = (float*)ws;                             // 32*2048*4   = 256 KB
    __bf16* Bp      = (__bf16*)(ws + (size_t)262144);         // 65536*2     = 128 KB
    float*  partial = (float*)(ws + (size_t)262144 + 131072); // 960*128*4   = 480 KB

    bcnn_prep<<<NB, 256, 0, stream>>>(feat, gF);
    bcnn_pack<<<65536 / 256, 256, 0, stream>>>(gF, Bp);
    bcnn_gemm<<<NWG, 128, 0, stream>>>(W, Bp, gF, partial);
    bcnn_finalize<<<(NB * NC + 255) / 256, 256, 0, stream>>>(partial, bias, out);
}
